// SeqRNNCRFModel_53266184405392
// MI455X (gfx1250) — matrix operand order repaired and checked by reading, not compiled, not run on hardware
//
#include <hip/hip_runtime.h>
#include <hip/hip_bf16.h>

// ---------------- problem constants (from reference setup_inputs) -----------
#define BB     256          // batch
#define TT     512          // time steps
#define KK     258          // tags + START + END
#define KP     288          // K padded: 288 = 9*32 (k-chunks) = 18*16 (n-tiles)
#define NKC    9            // k-chunks of 32
#define NT     18           // n-tiles of 16
#define STARTT 256
#define ENDD   257
#define NEG_INF_F (-1000000.0f)

typedef __attribute__((ext_vector_type(16))) _Float16 v16h;
typedef __attribute__((ext_vector_type(8)))  _Float16 v8h;
typedef __attribute__((ext_vector_type(8)))  float    v8f;

// ---------------- workspace layout (bytes) ----------------------------------
#define WS_OFF_W     0            // f16 [KP*KP] = 165888
#define WS_OFF_MROW  165888       // f32 [KP]    = 1152
#define WS_OFF_TERM  167040       // f32 [BB]    = 1024
#define WS_OFF_GOLD  168064       // f32 [BB]    = 1024

// ---------------- dynamic LDS layout (bytes) --------------------------------
#define L_OFF_W     0                               // f16 [KP*KP]   165888
#define L_OFF_FV    165888                          // f32 [16*KP]    18432
#define L_OFF_P     184320                          // f16 [16*KP]     9216
#define L_OFF_S     193536                          // f32 [16*KP]    18432
#define L_OFF_RED   211968                          // f32 [256]       1024
#define L_OFF_M     212992                          // f32 [16]          64
#define L_OFF_LEN   213056                          // i32 [16]          64
#define L_OFF_MROW  213120                          // f32 [KP]        1152
#define SMEM_BYTES  214272

// ---- fragment loaders (ISA §7.12.2 layouts) --------------------------------
// A 16x32 f16: lanes 0-15 = rows M, K runs {0-7,16-23}; lanes 16-31: {8-15,24-31}
__device__ __forceinline__ v16h load_a_frag(const _Float16* p, int lane, int kc) {
    int row = lane & 15;
    int k0  = kc * 32 + ((lane >> 4) << 3);
    const _Float16* ap = p + row * KP + k0;
    v8h lo = *(const v8h*)(ap);
    v8h hi = *(const v8h*)(ap + 16);
    return __builtin_shufflevector(lo, hi,
        0,1,2,3,4,5,6,7,8,9,10,11,12,13,14,15);
}
// B 32x16 f16: lanes 0-15 = cols N, K runs {0-7,16-23}; lanes 16-31: {8-15,24-31}
__device__ __forceinline__ v16h load_b_frag(const _Float16* w, int lane, int kc, int nt) {
    const _Float16* bp = w + (kc * 32 + ((lane >> 4) << 3)) * KP + nt * 16 + (lane & 15);
    v16h r;
#pragma unroll
    for (int s = 0; s < 8; ++s) {
        r[s]     = bp[s * KP];
        r[s + 8] = bp[(s + 16) * KP];
    }
    return r;
}

// ============================================================================
// Kernel 1: per-row max of transitions:  Mrow[j] = max_i trans[j,i]
// ============================================================================
__global__ void crf_rowmax(const float* __restrict__ trans, float* __restrict__ mrow) {
    int j = threadIdx.x;
    if (j >= KP) return;
    float m = NEG_INF_F;
    if (j < KK) {
        for (int i = 0; i < KK; ++i)
            m = fmaxf(m, trans[j * KK + i]);
        mrow[j] = m;
    } else {
        mrow[j] = 0.0f;
    }
}

// ============================================================================
// Kernel 2: W[i][j] = (f16) exp(trans[j,i] - Mrow[j]),  zero-padded to KP x KP
// ============================================================================
__global__ void crf_expw(const float* __restrict__ trans,
                         const float* __restrict__ mrow,
                         _Float16* __restrict__ w) {
    int idx = blockIdx.x * blockDim.x + threadIdx.x;
    if (idx >= KP * KP) return;
    int i = idx / KP;
    int j = idx - i * KP;
    float v = 0.0f;
    if (i < KK && j < KK)
        v = __expf(trans[j * KK + i] - mrow[j]);
    w[idx] = (_Float16)v;
}

// ============================================================================
// Kernel 3: forward recursion. 16 WGs x 256 threads (8 wave32s). W lives in
// LDS; per step: rowmax -> exp -> WMMA GEMM (16x288x288) -> log + emit.
// Each wave owns a PAIR of n-tiles (two independent accumulator chains) and
// B fragments are software-pipelined one k-chunk ahead.
// ============================================================================
__global__ void crf_forward(const float* __restrict__ emissions,
                            const float* __restrict__ trans,
                            const int*   __restrict__ lengths,
                            const _Float16* __restrict__ Wg,
                            const float* __restrict__ mrow_g,
                            float* __restrict__ term) {
    extern __shared__ __align__(16) char smem[];
    _Float16* w_lds   = (_Float16*)(smem + L_OFF_W);
    float*    fv      = (float*)   (smem + L_OFF_FV);
    _Float16* p_lds   = (_Float16*)(smem + L_OFF_P);
    float*    s_lds   = (float*)   (smem + L_OFF_S);
    float*    red     = (float*)   (smem + L_OFF_RED);
    float*    mloc    = (float*)   (smem + L_OFF_M);
    int*      len_l   = (int*)     (smem + L_OFF_LEN);
    float*    mrow_l  = (float*)   (smem + L_OFF_MROW);

    const int tid  = threadIdx.x;
    const int lane = tid & 31;
    const int wave = tid >> 5;
    const int b0   = blockIdx.x * 16;

    // ---- stage W (exp-transitions, f16) into LDS ----
    {
        const unsigned* src = (const unsigned*)Wg;
        unsigned*       dst = (unsigned*)w_lds;
        for (int i = tid; i < (KP * KP) / 2; i += 256) dst[i] = src[i];
        for (int i = tid; i < KP; i += 256) mrow_l[i] = mrow_g[i];
        if (tid < 16) len_l[tid] = lengths[b0 + tid];
    }
    // ---- init fv: NEG_INF except START column ----
    for (int idx = tid; idx < 16 * KP; idx += 256) {
        int j = idx % KP;
        fv[idx] = (j == STARTT) ? 0.0f : NEG_INF_F;
    }
    __syncthreads();

    for (int t = 1; t <= TT; ++t) {
        // ---- step 1: per-row max of fv (16 rows x 16 threads each) ----
        {
            int r = tid >> 4, c = tid & 15;
            float mx = NEG_INF_F;
            for (int j = c; j < KP; j += 16) mx = fmaxf(mx, fv[r * KP + j]);
            red[tid] = mx;
            __syncthreads();
            if (c == 0) {
                float m2 = red[r * 16];
#pragma unroll
                for (int q = 1; q < 16; ++q) m2 = fmaxf(m2, red[r * 16 + q]);
                mloc[r] = m2;
            }
            __syncthreads();
        }
        // ---- step 2: P = exp(fv - m) as f16 ----
        for (int idx = tid; idx < 16 * KP; idx += 256) {
            int r = idx / KP;
            p_lds[idx] = (_Float16)__expf(fv[idx] - mloc[r]);
        }
        __syncthreads();

        // ---- step 3: S = P @ W, two n-tiles per wave, pipelined B loads ----
        // A fragments are shared by all n-tiles: preload all 9 (72 VGPRs).
        {
            v16h afr[NKC];
#pragma unroll
            for (int kc = 0; kc < NKC; ++kc)
                afr[kc] = load_a_frag(p_lds, lane, kc);

            for (int base = 2 * wave; base < NT; base += 16) {
                const int nt0 = base, nt1 = base + 1;
                v8f acc0 = {}, acc1 = {};
                v16h bf0 = load_b_frag(w_lds, lane, 0, nt0);
                v16h bf1 = load_b_frag(w_lds, lane, 0, nt1);
#pragma unroll
                for (int kc = 0; kc < NKC; ++kc) {
                    v16h nb0 = bf0, nb1 = bf1;
                    if (kc + 1 < NKC) {           // prefetch next k-chunk
                        nb0 = load_b_frag(w_lds, lane, kc + 1, nt0);
                        nb1 = load_b_frag(w_lds, lane, kc + 1, nt1);
                    }
                    acc0 = __builtin_amdgcn_wmma_f32_16x16x32_f16(
                        false, afr[kc], false, bf0, (short)0, acc0, false, false);
                    acc1 = __builtin_amdgcn_wmma_f32_16x16x32_f16(
                        false, afr[kc], false, bf1, (short)0, acc1, false, false);
                    bf0 = nb0; bf1 = nb1;
                }
                // C layout: VGPR e -> M = e (+8 for upper half-wave), N = lane&15
                int n = lane & 15;
                int mb = (lane >> 4) << 3;
#pragma unroll
                for (int e = 0; e < 8; ++e) {
                    s_lds[(mb + e) * KP + nt0 * 16 + n] = acc0[e];
                    s_lds[(mb + e) * KP + nt1 * 16 + n] = acc1[e];
                }
            }
        }

        // ---- prefetch next step's emissions (global_prefetch_b8) ----
        if (t < TT) {
            const char* pf = (const char*)(emissions +
                ((size_t)(b0 + (tid >> 4)) * TT + t) * KK) + (tid & 15) * 64;
            __builtin_prefetch(pf, 0, 3);
        }
        __syncthreads();

        // ---- step 4: fv = where(t<=len, emit + m + Mrow + log S, fv) ----
        for (int idx = tid; idx < 16 * KP; idx += 256) {
            int r = idx / KP;
            int j = idx - r * KP;
            if (t <= len_l[r]) {
                float nv = NEG_INF_F;
                if (j < KK) {
                    float s = s_lds[idx];
                    if (s > 0.0f) {
                        float e = emissions[((size_t)(b0 + r) * TT + (t - 1)) * KK + j];
                        nv = mloc[r] + mrow_l[j] + e + __logf(s);
                    }
                }
                fv[idx] = nv;
            }
        }
        __syncthreads();
    }

    // ---- terminal: term[b] = logsumexp_j( fv[b,j] + trans[END,j] ) ----
    if (tid < 16) {
        const float* te = trans + (size_t)ENDD * KK;
        float mm = NEG_INF_F;
        for (int j = 0; j < KK; ++j) mm = fmaxf(mm, fv[tid * KP + j] + te[j]);
        float ss = 0.0f;
        for (int j = 0; j < KK; ++j) ss += __expf(fv[tid * KP + j] + te[j] - mm);
        term[b0 + tid] = mm + __logf(ss);
    }
}

// ============================================================================
// Kernel 4: gold path score per batch row
// ============================================================================
__global__ void crf_gold(const float* __restrict__ emissions,
                         const float* __restrict__ trans,
                         const int*   __restrict__ lengths,
                         const int*   __restrict__ tags,
                         float* __restrict__ gold) {
    __shared__ float red[256];
    int b = blockIdx.x, tid = threadIdx.x;
    int len = lengths[b];
    float acc = 0.0f;
    for (int t = tid; t < TT; t += 256) {
        if (t < len) {
            int nxt  = tags[b * TT + t];
            int prev = (t == 0) ? STARTT : tags[b * TT + t - 1];
            acc += trans[nxt * KK + prev] +
                   emissions[((size_t)b * TT + t) * KK + nxt];
        }
    }
    red[tid] = acc;
    __syncthreads();
    for (int s = 128; s > 0; s >>= 1) {
        if (tid < s) red[tid] += red[tid + s];
        __syncthreads();
    }
    if (tid == 0) {
        int last = tags[b * TT + len - 1];     // lengths >= 1
        gold[b] = red[0] + trans[ENDD * KK + last];
    }
}

// ============================================================================
// Kernel 5: loss = (sum term - sum gold) / B
// ============================================================================
__global__ void crf_final(const float* __restrict__ term,
                          const float* __restrict__ gold,
                          float* __restrict__ out) {
    __shared__ float red[256];
    int tid = threadIdx.x;
    red[tid] = term[tid] - gold[tid];
    __syncthreads();
    for (int s = 128; s > 0; s >>= 1) {
        if (tid < s) red[tid] += red[tid + s];
        __syncthreads();
    }
    if (tid == 0) out[0] = red[0] / (float)BB;
}

// ============================================================================
extern "C" void kernel_launch(void* const* d_in, const int* in_sizes, int n_in,
                              void* d_out, int out_size, void* d_ws, size_t ws_size,
                              hipStream_t stream) {
    const float* emissions = (const float*)d_in[0];  // (B,T,K) f32
    const float* trans     = (const float*)d_in[1];  // (K,K)   f32
    const int*   lengths   = (const int*)  d_in[2];  // (B,)    i32
    const int*   tags      = (const int*)  d_in[3];  // (B,T)   i32
    float*       out       = (float*)d_out;

    char* ws = (char*)d_ws;
    _Float16* W    = (_Float16*)(ws + WS_OFF_W);
    float*    Mrow = (float*)   (ws + WS_OFF_MROW);
    float*    term = (float*)   (ws + WS_OFF_TERM);
    float*    gold = (float*)   (ws + WS_OFF_GOLD);

    crf_rowmax<<<1, KP, 0, stream>>>(trans, Mrow);
    crf_expw<<<(KP * KP + 255) / 256, 256, 0, stream>>>(trans, Mrow, W);
    crf_forward<<<BB / 16, 256, SMEM_BYTES, stream>>>(emissions, trans, lengths,
                                                      W, Mrow, term);
    crf_gold<<<BB, 256, 0, stream>>>(emissions, trans, lengths, tags, gold);
    crf_final<<<1, 256, 0, stream>>>(term, gold, out);
}
